// IntBlockNd_73581379715454
// MI455X (gfx1250) — matrix operand order repaired and checked by reading, not compiled, not run on hardware
//
#include <hip/hip_runtime.h>

typedef __attribute__((ext_vector_type(16))) _Float16 v16h;
typedef __attribute__((ext_vector_type(8)))  _Float16 v8h;
typedef __attribute__((ext_vector_type(4)))  _Float16 v4h;
typedef __attribute__((ext_vector_type(8)))  float    v8f;
typedef int v4i __attribute__((vector_size(16)));   // matches builtin's param type

#define N_PTS 4096
#define CH    256
#define BATCH 8
#define JDIM  (BATCH * CH)      // 2048

// ---------------------------------------------------------------------------
// CDNA5 async global->LDS copy. Probe round confirmed the builtin exists and
// takes pointers-to-int4; param 1 is AS(1) ("__device__"), so order is
// (global src, lds dst, imm offset, imm cpol).
// ---------------------------------------------------------------------------
#if defined(__AMDGCN__) && __has_builtin(__builtin_amdgcn_global_load_async_to_lds_b128)
#define USE_ASYNC_LDS 1
#else
#define USE_ASYNC_LDS 0
#endif

__device__ __forceinline__ void stage_b128(const _Float16* g, _Float16* l) {
#if USE_ASYNC_LDS
    // generic->AS(1) is identity; generic->AS(3) is low 32 bits (ISA flat->LDS
    // truncation rule), so integer casts produce the right operands.
    __builtin_amdgcn_global_load_async_to_lds_b128(
        (__attribute__((address_space(1))) v4i*)(uintptr_t)g,
        (__attribute__((address_space(3))) v4i*)(uint32_t)(uintptr_t)l,
        0, 0);
#else
    *(uint4*)l = *(const uint4*)g;
#endif
}

__device__ __forceinline__ void stage_wait() {
#if USE_ASYNC_LDS
#if __has_builtin(__builtin_amdgcn_s_wait_asynccnt)
    __builtin_amdgcn_s_wait_asynccnt(0);
#else
    asm volatile("s_wait_asynccnt 0" ::: "memory");
#endif
#endif
}

// ---------------------------------------------------------------------------
// Stage 1: A[m,n] = w[m] * exp(-(t_m - t_n)^2 / (2 sigma^2)), stored f16
// ---------------------------------------------------------------------------
__global__ void build_A_kernel(const float* __restrict__ t,
                               const float* __restrict__ w,
                               const float* __restrict__ nu,
                               _Float16* __restrict__ A) {
    int idx = blockIdx.x * blockDim.x + threadIdx.x;   // N*N/4 threads
    int m  = idx >> 10;
    int n4 = (idx & 1023) << 2;
    float sigma = nu[0];
    float inv   = -0.5f / (sigma * sigma);
    float tm = t[m];
    float wm = w[m];
    v4h o;
#pragma unroll
    for (int k = 0; k < 4; ++k) {
        float d = tm - t[n4 + k];
        o[k] = (_Float16)(wm * __expf(d * d * inv));
    }
    *(v4h*)(A + (size_t)m * N_PTS + n4) = o;
}

// ---------------------------------------------------------------------------
// Stage 2: HT[j, n] = (f16) h_t[b, n, c], j = b*256 + c (LDS tile transpose)
// ---------------------------------------------------------------------------
__global__ void pack_HT_kernel(const float* __restrict__ h,
                               _Float16* __restrict__ HT) {
    __shared__ float s[32][33];
    int n0 = blockIdx.x * 32;
    int c0 = blockIdx.y * 32;
    int b  = blockIdx.z;
    int tx = threadIdx.x & 31;
    int ty = threadIdx.x >> 5;
    const float* src = h + (size_t)b * N_PTS * CH;
#pragma unroll
    for (int r = 0; r < 4; ++r) {
        int nn = ty + r * 8;
        s[nn][tx] = src[(size_t)(n0 + nn) * CH + (c0 + tx)];
    }
    __syncthreads();
#pragma unroll
    for (int r = 0; r < 4; ++r) {
        int cc = ty + r * 8;
        int j  = b * CH + c0 + cc;
        HT[(size_t)j * N_PTS + n0 + tx] = (_Float16)s[tx][cc];
    }
}

// ---------------------------------------------------------------------------
// Stage 3: OUT'[m, j] = sum_n A[m,n] * HT[j,n]  via v_wmma_f32_16x16x32_f16
// Block 128(M) x 256(J), K-step 32, 8 waves, each wave a 64x64 macro-tile
// (4x4 WMMA tiles -> 16 ds_load_b128 : 16 wmma per K-step), double-buffered
// LDS with one barrier per K-step.
// ---------------------------------------------------------------------------
#define BM 128
#define BJ 256
#define BK 32
#define KSTEPS (N_PTS / BK)   // 128
#define LDSW (BK + 8)         // 40 halfs row stride: 16B aligned + bank pad

__global__ __launch_bounds__(256) void gemm_kernel(const _Float16* __restrict__ A,
                                                   const _Float16* __restrict__ HT,
                                                   float* __restrict__ out) {
    __shared__ alignas(16) _Float16 As[2][BM][LDSW];   // 20 KB
    __shared__ alignas(16) _Float16 Bs[2][BJ][LDSW];   // 40 KB

    const int j0   = blockIdx.x * BJ;
    const int m0   = blockIdx.y * BM;
    const int tid  = threadIdx.x;
    const int lane = tid & 31;
    const int wave = tid >> 5;          // 0..7
    const int mW   = (wave >> 2) * 64;  // 0 / 64
    const int jW   = (wave & 3) * 64;   // 0 / 64 / 128 / 192

    // staging: per K-step each thread moves 16 halfs of A and 32 halfs of B
    const int aRow = tid >> 1;
    const int aSeg = (tid & 1) * 16;
    const _Float16* gA = A  + (size_t)(m0 + aRow) * N_PTS + aSeg;
    const _Float16* gB = HT + (size_t)(j0 + tid) * N_PTS;

    // fragment lane decode (ISA 7.12.2 16-bit layouts, wave32)
    const int fm = lane & 15;
    const int hg = lane >> 4;
    const int kA = hg * 8;
    const int kB = hg * 16;

    v8f acc[4][4] = {};

    auto stage = [&](int kt, int buf) {
        const _Float16* pa = gA + (size_t)kt * BK;
        stage_b128(pa,     &As[buf][aRow][aSeg]);
        stage_b128(pa + 8, &As[buf][aRow][aSeg + 8]);
        const _Float16* pb = gB + (size_t)kt * BK;
        stage_b128(pb,      &Bs[buf][tid][0]);
        stage_b128(pb + 8,  &Bs[buf][tid][8]);
        stage_b128(pb + 16, &Bs[buf][tid][16]);
        stage_b128(pb + 24, &Bs[buf][tid][24]);
    };

    stage(0, 0);
    stage_wait();
    __syncthreads();

    for (int kt = 0; kt < KSTEPS; ++kt) {
        const int cur = kt & 1;
        // fill the other buffer; previous barrier guarantees nobody is
        // still reading it
        if (kt + 1 < KSTEPS) stage(kt + 1, 1 - cur);

        // B fragments: column j, K halfs kA..kA+7 and 16+kA..16+kA+7 per lane
        v16h bf[4];
#pragma unroll
        for (int nt = 0; nt < 4; ++nt) {
            const _Float16* p = &Bs[cur][jW + nt * 16 + fm][kA];
            v8h lo = *(const v8h*)(p);
            v8h hi = *(const v8h*)(p + 16);
            bf[nt] = __builtin_shufflevector(lo, hi, 0,1,2,3,4,5,6,7,
                                                     8,9,10,11,12,13,14,15);
        }
        // A fragments + 16 WMMAs
#pragma unroll
        for (int mt = 0; mt < 4; ++mt) {
            const _Float16* p = &As[cur][mW + mt * 16 + fm][0];
            v8h lo = *(const v8h*)(p + kA);
            v8h hi = *(const v8h*)(p + 16 + kA);
            v16h af = __builtin_shufflevector(lo, hi, 0,1,2,3,4,5,6,7,
                                                      8,9,10,11,12,13,14,15);
#pragma unroll
            for (int nt = 0; nt < 4; ++nt) {
                acc[mt][nt] = __builtin_amdgcn_wmma_f32_16x16x32_f16(
                    false, af, false, bf[nt], (short)0, acc[mt][nt],
                    false, false);
            }
        }
        stage_wait();
        __syncthreads();
    }

    // epilogue: BJ == 256 -> whole block lies inside a single batch
    const int bb = j0 >> 8;
#pragma unroll
    for (int mt = 0; mt < 4; ++mt) {
#pragma unroll
        for (int nt = 0; nt < 4; ++nt) {
            int cc = jW + nt * 16 + fm;
            float* dst = out + (size_t)bb * N_PTS * CH + cc;
            int mb = m0 + mW + mt * 16 + hg * 8;   // acc element v -> M = mb+v
#pragma unroll
            for (int v = 0; v < 8; ++v) {
                dst[(size_t)(mb + v) * CH] = acc[mt][nt][v];
            }
        }
    }
}

// ---------------------------------------------------------------------------
extern "C" void kernel_launch(void* const* d_in, const int* in_sizes, int n_in,
                              void* d_out, int out_size, void* d_ws, size_t ws_size,
                              hipStream_t stream) {
    const float* h_t = (const float*)d_in[0];   // [B, N, C] f32
    const float* t   = (const float*)d_in[1];   // [N] f32
    const float* w   = (const float*)d_in[2];   // [N,1] f32
    const float* nu  = (const float*)d_in[3];   // [1] f32
    float* out = (float*)d_out;                 // [B, N, C] f32

    _Float16* Af = (_Float16*)d_ws;                     // 4096*4096 f16
    _Float16* HT = Af + (size_t)N_PTS * N_PTS;          // 2048*4096 f16

    build_A_kernel<<<(N_PTS * N_PTS / 4) / 256, 256, 0, stream>>>(t, w, nu, Af);
    pack_HT_kernel<<<dim3(N_PTS / 32, CH / 32, BATCH), 256, 0, stream>>>(h_t, HT);
    gemm_kernel<<<dim3(JDIM / BJ, N_PTS / BM), 256, 0, stream>>>(Af, HT, out);
}
